// MultiHeadAttentionLayer_2508260901339
// MI455X (gfx1250) — matrix operand order repaired and checked by reading, not compiled, not run on hardware
//
#include <hip/hip_runtime.h>
#include <hip/hip_bf16.h>

#define BATCH 4
#define SEQ   1024
#define HIDN  512
#define HEADS 8
#define HDIM  64
#define BL    (BATCH * SEQ)   // 4096
#define NREL  6

typedef __attribute__((ext_vector_type(16))) __bf16 v16bf;
typedef __attribute__((ext_vector_type(8)))  __bf16 bf16x8;
typedef __attribute__((ext_vector_type(8)))  float  v8f;

// native hardware converts (v_cvt_* on gfx1250) instead of bit-twiddled RNE
static __device__ __forceinline__ float  bf2f(__bf16 b) { return (float)b; }
static __device__ __forceinline__ __bf16 f2bf(float f)  { return (__bf16)f; }

// reductions across the 16-lane halves of a wave32 (xor of bits 0..3)
static __device__ __forceinline__ float half_max(float v) {
  #pragma unroll
  for (int off = 1; off < 16; off <<= 1) v = fmaxf(v, __shfl_xor(v, off, 32));
  return v;
}
static __device__ __forceinline__ float half_sum(float v) {
  #pragma unroll
  for (int off = 1; off < 16; off <<= 1) v += __shfl_xor(v, off, 32);
  return v;
}

// ---------------- fp32 -> bf16 elementwise ----------------
__global__ __launch_bounds__(256) void cvt_f32_bf16(const float* __restrict__ src,
                                                    __bf16* __restrict__ dst, int n) {
  int i = blockIdx.x * blockDim.x + threadIdx.x;
  if (i < n) dst[i] = f2bf(src[i]);
}

// ---------------- GEMM: C[m,n] = sum_k A[m,k] * W[n,k] + bias[n] ----------------
// A: [4096,512] fp32 (A_BF16=false) or bf16 (true). W: [512,512] bf16 row-major.
// MODE 0: bf16 out at [(b*H+h)*SEQ + l]*HDIM + d        (Q/K head layout)
// MODE 1: bf16 out at [(b*H+h)*HDIM + d]*SEQ + l        (V transposed)
// MODE 2: fp32 out at row*HIDN + col                    (final projection)
template <bool A_BF16, int MODE>
__global__ __launch_bounds__(256) void gemm_wmma(const void* __restrict__ Aptr,
                                                 const __bf16* __restrict__ W,
                                                 const float* __restrict__ bias,
                                                 void* __restrict__ Cptr) {
  const int lane = threadIdx.x & 31;
  const int wave = threadIdx.x >> 5;
  const int ln   = lane & 15;
  const int hi   = lane >> 4;
  const int c0   = hi * 8;
  const int m0   = blockIdx.x * 128 + wave * 16;
  const int n0   = blockIdx.y * 64;

  v8f zero = {};
  v8f acc[4];
  #pragma unroll
  for (int t = 0; t < 4; ++t) acc[t] = zero;

  for (int kb = 0; kb < HIDN; kb += 32) {
    // A fragment: lane row = m0+ln, K runs [c0,c0+8) and [16+c0,16+c0+8)
    v16bf a;
    if (A_BF16) {
      const __bf16* arow = (const __bf16*)Aptr + (size_t)(m0 + ln) * HIDN + kb;
      bf16x8 r0 = *(const bf16x8*)(arow + c0);
      bf16x8 r1 = *(const bf16x8*)(arow + 16 + c0);
      #pragma unroll
      for (int i = 0; i < 8; ++i) { a[i] = r0[i]; a[8 + i] = r1[i]; }
    } else {
      const float* arow = (const float*)Aptr + (size_t)(m0 + ln) * HIDN + kb;
      float tmp[16];
      *(float4*)(tmp + 0)  = *(const float4*)(arow + c0);
      *(float4*)(tmp + 4)  = *(const float4*)(arow + c0 + 4);
      *(float4*)(tmp + 8)  = *(const float4*)(arow + 16 + c0);
      *(float4*)(tmp + 12) = *(const float4*)(arow + 16 + c0 + 4);
      #pragma unroll
      for (int i = 0; i < 16; ++i) a[i] = f2bf(tmp[i]);
    }
    #pragma unroll
    for (int t = 0; t < 4; ++t) {
      // B fragment: lane col = n0+t*16+ln, K runs [hi*8,hi*8+8) and [16+hi*8,16+hi*8+8)
      const __bf16* wrow = W + (size_t)(n0 + t * 16 + ln) * HIDN + kb + hi * 8;
      bf16x8 b0 = *(const bf16x8*)wrow;
      bf16x8 b1 = *(const bf16x8*)(wrow + 16);
      v16bf bv;
      #pragma unroll
      for (int i = 0; i < 8; ++i) { bv[i] = b0[i]; bv[8 + i] = b1[i]; }
      acc[t] = __builtin_amdgcn_wmma_f32_16x16x32_bf16(false, a, false, bv,
                                                       (short)0, acc[t], false, false);
    }
  }

  #pragma unroll
  for (int t = 0; t < 4; ++t) {
    #pragma unroll
    for (int j = 0; j < 8; ++j) {
      const int row = m0 + j + hi * 8;
      const int col = n0 + t * 16 + ln;
      const float v = acc[t][j] + bias[col];
      if (MODE == 2) {
        ((float*)Cptr)[(size_t)row * HIDN + col] = v;
      } else {
        const int b = row >> 10, l = row & 1023, h = col >> 6, d = col & 63;
        const size_t idx = (MODE == 0)
            ? ((size_t)(b * HEADS + h) * SEQ + l) * HDIM + d
            : ((size_t)(b * HEADS + h) * HDIM + d) * SEQ + l;
        ((__bf16*)Cptr)[idx] = f2bf(v);
      }
    }
  }
}

// ---------------- s2[bh,q,j] = Q[bh,q,:] . rel_k[j,:] ----------------
__global__ __launch_bounds__(256) void relk_scores(const __bf16* __restrict__ Qb,
                                                   const float* __restrict__ rel_k,
                                                   float* __restrict__ s2) {
  const int t = blockIdx.x * blockDim.x + threadIdx.x; // bh*SEQ + q
  if (t >= BATCH * HEADS * SEQ) return;
  const __bf16* qrow = Qb + (size_t)t * HDIM;
  float qv[HDIM];
  #pragma unroll
  for (int i = 0; i < HDIM; ++i) qv[i] = bf2f(qrow[i]);
  #pragma unroll
  for (int j = 0; j < NREL; ++j) {
    float dot = 0.f;
    #pragma unroll
    for (int i = 0; i < HDIM; ++i) dot += qv[i] * rel_k[j * HDIM + i];
    s2[(size_t)t * NREL + j] = dot;
  }
}

// ---------------- causal flash attention with relative positions ----------------
__global__ __launch_bounds__(256) void attn_kernel(const __bf16* __restrict__ Qb,
                                                   const __bf16* __restrict__ Kb,
                                                   const __bf16* __restrict__ Vt,
                                                   const float* __restrict__ s2g,
                                                   const float* __restrict__ rel_v,
                                                   __bf16* __restrict__ xb) {
  __shared__ __bf16 Psh[8][16][32];  // per-wave P bounce, already bf16 (A-layout rows)
  __shared__ float s2sh[8][16][8];   // per-wave s2 rows (padded 6->8)
  __shared__ float msh[8][16], lsh[8][16];
  __shared__ float w2sh[8][16][64];  // rel_v contribution per row

  const int lane = threadIdx.x & 31;
  const int wave = threadIdx.x >> 5;
  const int ln   = lane & 15;
  const int hi   = lane >> 4;
  const int c0   = hi * 8;
  const int bh   = blockIdx.y;                    // b*HEADS + h
  const int qb   = blockIdx.x * 128 + wave * 16;  // this wave's 16 query rows

  if (lane < 16) {
    const float* sp = s2g + (size_t)(bh * SEQ + qb + lane) * NREL;
    #pragma unroll
    for (int j = 0; j < NREL; ++j) s2sh[wave][lane][j] = sp[j];
  }
  asm volatile("s_wait_dscnt 0x0" ::: "memory");

  // Q A-fragments for d=[0,32) and [32,64)
  const __bf16* qrow = Qb + (size_t)(bh * SEQ + qb + ln) * HDIM;
  v16bf aQ[2];
  #pragma unroll
  for (int f = 0; f < 2; ++f) {
    bf16x8 r0 = *(const bf16x8*)(qrow + f * 32 + c0);
    bf16x8 r1 = *(const bf16x8*)(qrow + f * 32 + 16 + c0);
    #pragma unroll
    for (int i = 0; i < 8; ++i) { aQ[f][i] = r0[i]; aQ[f][8 + i] = r1[i]; }
  }

  v8f zero = {};
  v8f O[4];
  #pragma unroll
  for (int t = 0; t < 4; ++t) O[t] = zero;
  float m[8], l[8];
  #pragma unroll
  for (int j = 0; j < 8; ++j) { m[j] = -1e30f; l[j] = 0.f; }

  const int nb = (qb + 16 + 31) >> 5;  // 32-key blocks (keys <= qb+15)
  for (int blk = 0; blk < nb; ++blk) {
    const int kb = blk * 32;
    // ---- scores: two 16x16 tiles, each = two chained WMMAs over d ----
    v8f S[2];
    #pragma unroll
    for (int t2 = 0; t2 < 2; ++t2) {
      const __bf16* krow = Kb + (size_t)(bh * SEQ + kb + t2 * 16 + ln) * HDIM + hi * 8;
      bf16x8 x0 = *(const bf16x8*)(krow);
      bf16x8 x1 = *(const bf16x8*)(krow + 16);
      bf16x8 y0 = *(const bf16x8*)(krow + 32);
      bf16x8 y1 = *(const bf16x8*)(krow + 48);
      v16bf b0, b1;
      #pragma unroll
      for (int i = 0; i < 8; ++i) {
        b0[i] = x0[i]; b0[8 + i] = x1[i];
        b1[i] = y0[i]; b1[8 + i] = y1[i];
      }
      v8f c = zero;
      c = __builtin_amdgcn_wmma_f32_16x16x32_bf16(false, aQ[0], false, b0, (short)0, c, false, false);
      c = __builtin_amdgcn_wmma_f32_16x16x32_bf16(false, aQ[1], false, b1, (short)0, c, false, false);
      S[t2] = c;
    }
    // ---- online softmax with rel-k bias + causal mask ----
    float alpha[8];
    #pragma unroll
    for (int j = 0; j < 8; ++j) {
      const int rl = j + hi * 8;
      const int q  = qb + rl;
      const int k0 = kb + ln, k1 = kb + 16 + ln;
      const int i0 = min(max(k0 - q + 5, 0), 5);
      const int i1 = min(max(k1 - q + 5, 0), 5);
      float s0 = (S[0][j] + s2sh[wave][rl][i0]) * 0.125f;
      float s1 = (S[1][j] + s2sh[wave][rl][i1]) * 0.125f;
      if (k0 > q) s0 = -1e30f;
      if (k1 > q) s1 = -1e30f;
      const float rm = half_max(fmaxf(s0, s1));
      const float mn = fmaxf(m[j], rm);
      const float a  = __expf(m[j] - mn);
      const float p0 = __expf(s0 - mn);
      const float p1 = __expf(s1 - mn);
      l[j] = l[j] * a + half_sum(p0 + p1);
      m[j] = mn; alpha[j] = a;
      Psh[wave][rl][ln]      = f2bf(p0);
      Psh[wave][rl][16 + ln] = f2bf(p1);
    }
    #pragma unroll
    for (int t = 0; t < 4; ++t)
      #pragma unroll
      for (int j = 0; j < 8; ++j) O[t][j] *= alpha[j];
    asm volatile("s_wait_dscnt 0x0" ::: "memory");
    // ---- P (16x32) as A-fragment straight from LDS (bf16 runs) ----
    v16bf aP;
    {
      const __bf16* prow = &Psh[wave][ln][0];
      bf16x8 r0 = *(const bf16x8*)(prow + c0);
      bf16x8 r1 = *(const bf16x8*)(prow + 16 + c0);
      #pragma unroll
      for (int i = 0; i < 8; ++i) { aP[i] = r0[i]; aP[8 + i] = r1[i]; }
    }
    // ---- O += P @ V  (V transposed: contiguous key runs) ----
    #pragma unroll
    for (int t = 0; t < 4; ++t) {
      const __bf16* vrow = Vt + (size_t)(bh * HDIM + t * 16 + ln) * SEQ + kb + hi * 8;
      bf16x8 v0 = *(const bf16x8*)vrow;
      bf16x8 v1 = *(const bf16x8*)(vrow + 16);
      v16bf bv;
      #pragma unroll
      for (int i = 0; i < 8; ++i) { bv[i] = v0[i]; bv[8 + i] = v1[i]; }
      O[t] = __builtin_amdgcn_wmma_f32_16x16x32_bf16(false, aP, false, bv, (short)0, O[t], false, false);
    }
  }

  // ---- publish per-row m, l ----
  if (ln == 0) {
    #pragma unroll
    for (int j = 0; j < 8; ++j) { msh[wave][j + hi * 8] = m[j]; lsh[wave][j + hi * 8] = l[j]; }
  }
  asm volatile("s_wait_dscnt 0x0" ::: "memory");

  // ---- rel_v term: c5..c1 = attn on the 5-wide diagonal band, c0 = 1 - sum ----
  if (lane < 16) {
    const int q = qb + lane;
    const float mF = msh[wave][lane];
    const float lF = lsh[wave][lane];
    const __bf16* qr = Qb + (size_t)(bh * SEQ + q) * HDIM;
    float c[6] = {0.f, 0.f, 0.f, 0.f, 0.f, 0.f};
    float csum = 0.f;
    #pragma unroll
    for (int t = 0; t < 5; ++t) {
      const int k = q - 4 + t;
      if (k >= 0) {
        const __bf16* kr = Kb + (size_t)(bh * SEQ + k) * HDIM;
        float dot = 0.f;
        #pragma unroll 8
        for (int i = 0; i < HDIM; ++i) dot += bf2f(qr[i]) * bf2f(kr[i]);
        const float s = (dot + s2sh[wave][lane][t + 1]) * 0.125f;
        const float p = __expf(s - mF) / lF;
        c[t + 1] = p; csum += p;
      }
    }
    c[0] = 1.f - csum;
    for (int d = 0; d < HDIM; ++d) {
      float w = 0.f;
      #pragma unroll
      for (int j = 0; j < NREL; ++j) w += c[j] * rel_v[j * HDIM + d];
      w2sh[wave][lane][d] = w;
    }
  }
  asm volatile("s_wait_dscnt 0x0" ::: "memory");

  // ---- finalize and write [B, L, HIDN] bf16 ----
  const int b = bh >> 3, h = bh & 7;
  #pragma unroll
  for (int t = 0; t < 4; ++t) {
    #pragma unroll
    for (int j = 0; j < 8; ++j) {
      const int rl = j + hi * 8;
      const int q  = qb + rl;
      const int d  = t * 16 + ln;
      const float v = O[t][j] / l[j] + w2sh[wave][rl][d];
      xb[((size_t)(b * SEQ + q)) * HIDN + h * HDIM + d] = f2bf(v);
    }
  }
}

extern "C" void kernel_launch(void* const* d_in, const int* in_sizes, int n_in,
                              void* d_out, int out_size, void* d_ws, size_t ws_size,
                              hipStream_t stream) {
  (void)in_sizes; (void)n_in; (void)out_size; (void)ws_size;
  const float* query = (const float*)d_in[0];
  const float* key   = (const float*)d_in[1];
  const float* value = (const float*)d_in[2];
  const float* Wq    = (const float*)d_in[3];
  const float* bq    = (const float*)d_in[4];
  const float* Wk    = (const float*)d_in[5];
  const float* bk    = (const float*)d_in[6];
  const float* Wv    = (const float*)d_in[7];
  const float* bv    = (const float*)d_in[8];
  const float* Wo    = (const float*)d_in[9];
  const float* bo    = (const float*)d_in[10];
  const float* rel_k = (const float*)d_in[11];
  const float* rel_v = (const float*)d_in[12];

  // workspace layout (~19 MB)
  char* p = (char*)d_ws;
  __bf16* Wqb = (__bf16*)p; p += (size_t)HIDN * HIDN * 2;
  __bf16* Wkb = (__bf16*)p; p += (size_t)HIDN * HIDN * 2;
  __bf16* Wvb = (__bf16*)p; p += (size_t)HIDN * HIDN * 2;
  __bf16* Wob = (__bf16*)p; p += (size_t)HIDN * HIDN * 2;
  __bf16* Qb  = (__bf16*)p; p += (size_t)BL * HIDN * 2;    // [B,H,L,64]
  __bf16* Kbf = (__bf16*)p; p += (size_t)BL * HIDN * 2;    // [B,H,L,64]
  __bf16* Vt  = (__bf16*)p; p += (size_t)BL * HIDN * 2;    // [B,H,64,L]
  float*  s2  = (float*)p;  p += (size_t)BATCH * HEADS * SEQ * NREL * 4;
  __bf16* xb  = (__bf16*)p; p += (size_t)BL * HIDN * 2;    // [B,L,512]

  const int NW = HIDN * HIDN;
  cvt_f32_bf16<<<(NW + 255) / 256, 256, 0, stream>>>(Wq, Wqb, NW);
  cvt_f32_bf16<<<(NW + 255) / 256, 256, 0, stream>>>(Wk, Wkb, NW);
  cvt_f32_bf16<<<(NW + 255) / 256, 256, 0, stream>>>(Wv, Wvb, NW);
  cvt_f32_bf16<<<(NW + 255) / 256, 256, 0, stream>>>(Wo, Wob, NW);

  dim3 ggrid(BL / 128, HIDN / 64);
  gemm_wmma<false, 0><<<ggrid, 256, 0, stream>>>(query, Wqb, bq, Qb);
  gemm_wmma<false, 0><<<ggrid, 256, 0, stream>>>(key,   Wkb, bk, Kbf);
  gemm_wmma<false, 1><<<ggrid, 256, 0, stream>>>(value, Wvb, bv, Vt);

  relk_scores<<<(BATCH * HEADS * SEQ) / 256, 256, 0, stream>>>(Qb, rel_k, s2);

  attn_kernel<<<dim3(SEQ / 128, BATCH * HEADS), 256, 0, stream>>>(Qb, Kbf, Vt, s2, rel_v, xb);

  gemm_wmma<true, 2><<<ggrid, 256, 0, stream>>>(xb, Wob, bo, (float*)d_out);
}
